// MyModel_82944408421020
// MI455X (gfx1250) — matrix operand order repaired and checked by reading, not compiled, not run on hardware
//
#include <hip/hip_runtime.h>
#include <hip/hip_bf16.h>
#include <stdint.h>

// Problem dims (fixed by reference)
#define EMBD   256
#define HIDD   256
#define GATES  1024          // 4*HID
#define NROWS  32768         // T*B rows (= sequential scan steps)
#define NSTEP  32768
#define BATCH  64
#define LROW   260           // LDS row stride in floats (256 + 4 pad)

typedef __bf16 bf16t;
typedef bf16t v16bf __attribute__((ext_vector_type(16)));
typedef bf16t v8bf  __attribute__((ext_vector_type(8)));
typedef float v8f   __attribute__((ext_vector_type(8)));
typedef float v4f   __attribute__((ext_vector_type(4)));
typedef unsigned int u32;
typedef u32 v4u __attribute__((ext_vector_type(4)));
typedef int  v8i __attribute__((ext_vector_type(8)));
typedef int  v4i __attribute__((ext_vector_type(4)));

// Workspace layout (bytes)
#define OFF_H    0                         // 2*256 f32 h double buffer
#define OFF_PRE  4096                      // 32768*1024 f32   (128 MB)
#define OFF_XEH  (4096 + 134217728)        // 32768*256 bf16   (16 MB)
#define OFF_XEL  (OFF_XEH + 16777216)      // 32768*256 bf16   (16 MB)
#define OFF_WH   (OFF_XEL + 16777216)      // 1024*256 bf16    (0.5 MB)
#define OFF_WL   (OFF_WH  + 524288)        // 1024*256 bf16    (0.5 MB)

// ---------------------------------------------------------------------------
// Precompute 0: split W_ih into bf16 hi/lo planes (one block per row).
// ---------------------------------------------------------------------------
__global__ __launch_bounds__(256) void split_wih(
    const float* __restrict__ Wih, bf16t* __restrict__ wh, bf16t* __restrict__ wl)
{
    const long i = (long)blockIdx.x * EMBD + threadIdx.x;
    const float v = Wih[i];
    const bf16t h = (bf16t)v;
    wh[i] = h;
    wl[i] = (bf16t)(v - (float)h);
}

// ---------------------------------------------------------------------------
// Precompute 1: gather embedding rows and split into bf16 hi/lo planes.
// One block per output row m (token x[m]).
// ---------------------------------------------------------------------------
__global__ __launch_bounds__(256) void gather_split_emb(
    const int* __restrict__ x, const float* __restrict__ emb,
    bf16t* __restrict__ xh, bf16t* __restrict__ xl)
{
    const int  m   = blockIdx.x;
    const long tok = x[m];
    const float v  = emb[tok * (long)EMBD + threadIdx.x];
    const bf16t h  = (bf16t)v;
    const long o   = (long)m * EMBD + threadIdx.x;
    xh[o] = h;
    xl[o] = (bf16t)(v - (float)h);
}

// ---------------------------------------------------------------------------
// Phase 1 GEMM: pre[m,n] = xe[m,:] . W_ih[n,:] + (b_ih[n]+b_hh[n])
// One wave per 16(M) x 64(N) tile; operands pre-split to bf16 hi/lo, so the
// inner loop is pure b128 loads + 12 v_wmma per K-chunk (hi*hi + lo*hi into
// accA, hi*lo into accB -> no back-to-back WMMA RAW chain).
// ---------------------------------------------------------------------------
__global__ __launch_bounds__(256) void lstm_pre_wmma(
    const bf16t* __restrict__ xh, const bf16t* __restrict__ xl,
    const bf16t* __restrict__ wh, const bf16t* __restrict__ wl,
    const float* __restrict__ bih, const float* __restrict__ bhh,
    float* __restrict__ pre)
{
    const int lane = threadIdx.x & 31;
    const int wave = threadIdx.x >> 5;
    const int tile = blockIdx.x * 8 + wave;     // 32768 wave-tiles
    const int mt   = tile >> 4;                 // 2048 M-tiles
    const int nq   = tile & 15;                 // 16 groups of 4 N-tiles
    const int l16  = lane & 15;
    const int hi   = lane >> 4;                 // half-wave select

    const int m = mt * 16 + l16;
    const bf16t* ah_row = xh + (long)m * EMBD;
    const bf16t* al_row = xl + (long)m * EMBD;

    int ncol[4];
    #pragma unroll
    for (int j = 0; j < 4; ++j) ncol[j] = (nq * 4 + j) * 16 + l16;

    const v8f vzero = {0.f,0.f,0.f,0.f,0.f,0.f,0.f,0.f};
    v8f accA[4], accB[4];
    #pragma unroll
    for (int j = 0; j < 4; ++j) { accA[j] = vzero; accB[j] = vzero; }

    #pragma unroll
    for (int kc = 0; kc < EMBD; kc += 32) {
        // A k order used here (contiguous, matches B): lanes<16 carry K 0..15, lanes>=16 carry K 16..31
        const int ka = kc + hi * 16;
        v8bf a0h = *(const v8bf*)(ah_row + ka);
        v8bf a1h = *(const v8bf*)(ah_row + ka + 8);
        v8bf a0l = *(const v8bf*)(al_row + ka);
        v8bf a1l = *(const v8bf*)(al_row + ka + 8);
        v16bf ah = __builtin_shufflevector(a0h, a1h, 0,1,2,3,4,5,6,7,8,9,10,11,12,13,14,15);
        v16bf al = __builtin_shufflevector(a0l, a1l, 0,1,2,3,4,5,6,7,8,9,10,11,12,13,14,15);

        #pragma unroll
        for (int j = 0; j < 4; ++j) {
            // B k order used here (contiguous, matches A): lanes<16 carry K 0..15, lanes>=16 carry K 16..31
            const long kb = (long)ncol[j] * EMBD + kc + hi * 16;
            v16bf bh = *(const v16bf*)(wh + kb);
            v16bf bl = *(const v16bf*)(wl + kb);
            accA[j] = __builtin_amdgcn_wmma_f32_16x16x32_bf16(false, ah, false, bh, (short)0, accA[j], false, false);
            accB[j] = __builtin_amdgcn_wmma_f32_16x16x32_bf16(false, ah, false, bl, (short)0, accB[j], false, false);
            accA[j] = __builtin_amdgcn_wmma_f32_16x16x32_bf16(false, al, false, bh, (short)0, accA[j], false, false);
        }
    }

    // C/D layout: vgpr r, lanes<16 -> M=r, lanes>=16 -> M=8+r; N = lane%16
    #pragma unroll
    for (int j = 0; j < 4; ++j) {
        const float bias = bih[ncol[j]] + bhh[ncol[j]];
        #pragma unroll
        for (int r = 0; r < 8; ++r) {
            const int mrow = mt*16 + hi*8 + r;
            pre[(long)mrow * GATES + ncol[j]] = accA[j][r] + accB[j][r] + bias;
        }
    }
}

// ---------------------------------------------------------------------------
// Phase 2: sequential scan. 4 workgroups (one cluster) each own hidden dims
// [64*wg, 64*wg+64): W_hh rows {g*256 + 64*wg + d} resident in LDS via TDM
// (row-padded by the TDM pad feature). Per step: 256 threads x 256-MAC dot
// (4 independent FMA chains), gate nonlinearity for t<64, h exchanged via a
// double-buffered L2 slot + cluster barrier (-3). pre[] operand pipelined one
// step ahead and prefetched 8 steps ahead.
// ---------------------------------------------------------------------------
__device__ __forceinline__ float sigm(float v) { return 1.0f / (1.0f + __expf(-v)); }

__global__ __launch_bounds__(256) void lstm_seq(
    const float* __restrict__ pre, const float* __restrict__ Whh,
    float* __restrict__ hglob, float* __restrict__ out)
{
    extern __shared__ float smem[];
    float* Wl = smem;                  // 256 * LROW floats (W slice, padded rows)
    float* hl = smem + 256 * LROW;     // 256 floats: current h
    float* gl = hl + HIDD;             // 256 floats: gate pre-activations
    float* cl = gl + 256;              // 64 floats: local c slice

    const int t  = threadIdx.x;        // 0..255
    const int wg = blockIdx.x;         // 0..3 (cluster member)
    const int g  = t >> 6;             // gate index of this thread's row
    const int d  = t & 63;             // local hidden dim

    // ---- Stage W_hh slice into LDS via Tensor Data Mover ----
#if __has_builtin(__builtin_amdgcn_tensor_load_to_lds)
    if (t < 32) {                      // one wave issues the 4 TDM descriptors
        const u32 ldsbase = __builtin_amdgcn_groupstaticsize();
        #pragma unroll
        for (int gg = 0; gg < 4; ++gg) {
            const uint64_t ga = (uint64_t)(uintptr_t)(Whh + ((long)(256*gg + 64*wg)) * HIDD);
            // D# group0: count=1, lds_addr, 57b global_addr, type=2
            v4u g0 = { 1u,
                       ldsbase + (u32)(gg * 64 * LROW * 4),
                       (u32)ga,
                       ((u32)(ga >> 32) & 0x01FFFFFFu) | (2u << 30) };
            // D# group1: data_size=4B, pad_enable, pad 4 dwords every 256 dwords,
            // tensor_dim0=256, tensor_dim1=64, tile 256x64, dim0_stride=256
            v8i g1 = { (int)((2u<<16) | (1u<<20) | (7u<<22) | (3u<<25)),
                       (int)(256u << 16),
                       (int)(64u  << 16),
                       (int)(256u << 16),
                       64,
                       256,
                       0, 0 };
            v4i gz = {0,0,0,0};
#if defined(__clang_major__) && (__clang_major__ >= 23)
            v8i gz8 = {0,0,0,0,0,0,0,0};
            __builtin_amdgcn_tensor_load_to_lds(g0, g1, gz, gz, gz8, 0);
#else
            __builtin_amdgcn_tensor_load_to_lds(g0, g1, gz, gz, 0);
#endif
        }
        __builtin_amdgcn_s_wait_tensorcnt(0);
    }
#else
    for (int idx = t; idx < 256 * HIDD; idx += 256) {
        const int r = idx >> 8, k = idx & 255;
        Wl[r * LROW + k] = Whh[((long)(256*(r>>6) + 64*wg + (r&63))) * HIDD + k];
    }
#endif

    hl[t] = 0.0f;
    if (t < 64) cl[t] = 0.0f;
    __syncthreads();

    const float* prept = pre + 256*g + 64*wg + d;   // this thread's gate row in pre
    const v4f* w4 = (const v4f*)(Wl + t * LROW);    // 16B aligned (1040*t)
    const v4f* h4 = (const v4f*)hl;

    float pv = prept[0];                            // pipelined pre operand

    for (int n = 0; n < NSTEP; ++n) {
        // gate_row = pre[n, row] + W_hh[row,:] . h   (4 independent FMA chains)
        float a0 = 0.f, a1 = 0.f, a2 = 0.f, a3 = 0.f;
        #pragma unroll 16
        for (int k = 0; k < HIDD/4; ++k) {
            v4f w = w4[k]; v4f h = h4[k];
            a0 += w[0]*h[0]; a1 += w[1]*h[1]; a2 += w[2]*h[2]; a3 += w[3]*h[3];
        }
        gl[t] = pv + (a0 + a1) + (a2 + a3);
        __syncthreads();                            // gates visible in WG

        const int slot = n & 1;                     // double-buffered h slot
        if (t < 64) {
            const float iv = sigm(gl[t]);
            const float fv = sigm(gl[64 + t]);
            const float gv = tanhf(gl[128 + t]);
            const float ov = sigm(gl[192 + t]);
            const float c  = fv * cl[t] + iv * gv;
            cl[t] = c;
            const float h  = ov * tanhf(c);
            __hip_atomic_store(&hglob[slot*HIDD + 64*wg + t], h,
                               __ATOMIC_RELAXED, __HIP_MEMORY_SCOPE_AGENT);
            if (n >= NSTEP - BATCH)
                out[2*HIDD + (n - (NSTEP - BATCH))*HIDD + 64*wg + t] = h;
            if (n == NSTEP - 1) {
                out[64*wg + t]        = h;          // h_n
                out[HIDD + 64*wg + t] = c;          // c_n
            }
        }
        // own stores device-visible, WG-converged, then cluster-wide sync
        asm volatile("s_wait_storecnt 0" ::: "memory");
        __syncthreads();
        if (t < 32) asm volatile("s_barrier_signal -3" ::: "memory");
        asm volatile("s_barrier_wait -3" ::: "memory");
        // issue next step's pre load early; prefetch the stream 8 steps out
        if (n + 1 < NSTEP) pv = prept[(long)(n + 1) * GATES];
        if (n + 8 < NSTEP)
            __builtin_prefetch((const void*)(prept + (long)(n + 8) * GATES), 0, 0);
        // pull full h (all 4 slices) for next step
        hl[t] = __hip_atomic_load(&hglob[slot*HIDD + t],
                                  __ATOMIC_RELAXED, __HIP_MEMORY_SCOPE_AGENT);
        __syncthreads();
    }
}

// ---------------------------------------------------------------------------
extern "C" void kernel_launch(void* const* d_in, const int* in_sizes, int n_in,
                              void* d_out, int out_size, void* d_ws, size_t ws_size,
                              hipStream_t stream) {
    (void)in_sizes; (void)n_in; (void)out_size; (void)ws_size;
    const int*   x   = (const int*)  d_in[0];
    const float* emb = (const float*)d_in[1];
    const float* Wih = (const float*)d_in[2];
    const float* Whh = (const float*)d_in[3];
    const float* bih = (const float*)d_in[4];
    const float* bhh = (const float*)d_in[5];
    float* out   = (float*)d_out;

    char* ws = (char*)d_ws;
    float* hglob = (float*)(ws + OFF_H);
    float* pre   = (float*)(ws + OFF_PRE);
    bf16t* xh    = (bf16t*)(ws + OFF_XEH);
    bf16t* xl    = (bf16t*)(ws + OFF_XEL);
    bf16t* wh    = (bf16t*)(ws + OFF_WH);
    bf16t* wl    = (bf16t*)(ws + OFF_WL);

    // Precompute: bf16 hi/lo planes (gather fused into the emb split)
    split_wih<<<dim3(GATES), dim3(256), 0, stream>>>(Wih, wh, wl);
    gather_split_emb<<<dim3(NROWS), dim3(256), 0, stream>>>(x, emb, xh, xl);

    // Phase 1: 32768 wave-tiles (16x64 each) / 8 waves per block
    lstm_pre_wmma<<<dim3(4096), dim3(256), 0, stream>>>(xh, xl, wh, wl, bih, bhh, pre);

    // Phase 2: 4 workgroups (one cluster of 4 WGPs), ~268.5 KB LDS each
    const size_t lds_bytes = (size_t)(256*LROW + HIDD + 256 + 64) * sizeof(float);
    (void)hipFuncSetAttribute((const void*)lstm_seq,
                              hipFuncAttributeMaxDynamicSharedMemorySize,
                              (int)lds_bytes);
    lstm_seq<<<dim3(4), dim3(256), lds_bytes, stream>>>(pre, Whh, hglob, out);
}
